// TriangleSelfAttention_13786845020548
// MI455X (gfx1250) — matrix operand order repaired and checked by reading, not compiled, not run on hardware
//
#include <hip/hip_runtime.h>

#define NTOK 384
#define CCH  128
#define TTOK (NTOK * NTOK)
#define SCALE 0.08838834764831845f
#define EPSLN 1e-5f

typedef __attribute__((ext_vector_type(16))) _Float16 v16h;
typedef __attribute__((ext_vector_type(8)))  _Float16 v8h;
typedef __attribute__((ext_vector_type(8)))  float    v8f;

union HF16 { v16h v; v8h h[2]; _Float16 e[16]; };

// A-fragment (16x32 f16, M = lane%16). Per ISA 7.12.2: lane L holds
// K = (h<8?0:16) + (L/16)*8 + h%8  -> two contiguous 16B chunks at +0 and +16.
static __device__ inline v16h ldA(const _Float16* p, int lane, int stride) {
  const _Float16* r = p + (lane & 15) * stride + ((lane >> 4) << 3);
  HF16 a; a.h[0] = *(const v8h*)r; a.h[1] = *(const v8h*)(r + 16); return a.v;
}
// B-fragment (32x16 f16, N = lane%16, memory rows are the N dimension,
// K contiguous within a row): lane L holds K = (h<8?0:16) + (L/16)*8 + h%8  -> two 16B chunks at +0 and +16.
static __device__ inline v16h ldB(const _Float16* p, int lane, int stride) {
  const _Float16* r = p + (lane & 15) * stride + ((lane >> 4) << 3);
  HF16 b; b.h[0] = *(const v8h*)r; b.h[1] = *(const v8h*)(r + 16); return b.v;
}
// Pre-packed B-fragment: 32 lanes x 16 halves contiguous per tile.
static __device__ inline v16h ldBpk(const _Float16* p, int lane) {
  const _Float16* r = p + lane * 16;
  HF16 b; b.h[0] = *(const v8h*)r; b.h[1] = *(const v8h*)(r + 8); return b.v;
}
static __device__ inline v8f wmma16(v16h a, v16h b, v8f c) {
  return __builtin_amdgcn_wmma_f32_16x16x32_f16(false, a, false, b, (short)0, c, false, false);
}

// ---------------------------------------------------------------------------
// Kernel 1: repack Wq,Wk,Wv,Wg,Wo (each 128x128 f32, K-major rows) into
// B-fragment order: [mat][kTile 0..3][nTile 0..7][lane 0..31][half 0..15] f16.
// ---------------------------------------------------------------------------
__global__ __launch_bounds__(256) void k_pack(const float* __restrict__ Wq,
                                              const float* __restrict__ Wk,
                                              const float* __restrict__ Wv,
                                              const float* __restrict__ Wg,
                                              const float* __restrict__ Wo,
                                              _Float16* __restrict__ wPack) {
  int idx = blockIdx.x * 256 + threadIdx.x;
  if (idx >= 5 * 32 * 32 * 16) return;
  int h = idx & 15, lane = (idx >> 4) & 31, tile = idx >> 9;
  int nt = tile & 7, kb = (tile >> 3) & 3, m = tile >> 5;
  int row = kb * 32 + ((h >> 3) << 4) + ((lane >> 4) << 3) + (h & 7);   // K index
  int col = nt * 16 + (lane & 15);              // N index
  const float* W = (m == 0) ? Wq : (m == 1) ? Wk : (m == 2) ? Wv : (m == 3) ? Wg : Wo;
  wPack[idx] = (_Float16)W[row * 128 + col];
}

// ---------------------------------------------------------------------------
// Kernel 2: LayerNorm + projections q,k,v,g (WMMA) + bias vector b.
// 64 tokens per block, 256 threads (8 waves). v stored transposed vT[i][c][k].
// A-fragments are shared across all four weight matrices per tile; all
// epilogues are fully unrolled (no movrel / no per-element branching).
// ---------------------------------------------------------------------------
__global__ __launch_bounds__(256) void k_lnproj(const float* __restrict__ x,
                                                const float* __restrict__ lng,
                                                const float* __restrict__ lnb,
                                                const float* __restrict__ Wb,
                                                const float* __restrict__ bg,
                                                const _Float16* __restrict__ wPack,
                                                _Float16* __restrict__ qOut,
                                                _Float16* __restrict__ kOut,
                                                _Float16* __restrict__ gOut,
                                                _Float16* __restrict__ vTOut,
                                                float* __restrict__ bOut) {
  __shared__ _Float16 nT[64 * 128];
  __shared__ float bp[64 * 4];
  const int lane = threadIdx.x & 31, w = threadIdx.x >> 5;
  const int t0 = blockIdx.x * 64;

  float4 lg4 = *(const float4*)&lng[lane * 4];
  float4 lb4 = *(const float4*)&lnb[lane * 4];
#pragma unroll
  for (int tt = 0; tt < 8; ++tt) {
    int tl = w * 8 + tt;
    float4 xv = *(const float4*)&x[((size_t)(t0 + tl)) * 128 + lane * 4];
    float s = xv.x + xv.y + xv.z + xv.w;
#pragma unroll
    for (int msk = 16; msk; msk >>= 1) s += __shfl_xor(s, msk);
    float mu = s * (1.f / 128.f);
    float d0 = xv.x - mu, d1 = xv.y - mu, d2 = xv.z - mu, d3 = xv.w - mu;
    float ss = d0 * d0 + d1 * d1 + d2 * d2 + d3 * d3;
#pragma unroll
    for (int msk = 16; msk; msk >>= 1) ss += __shfl_xor(ss, msk);
    float rs = rsqrtf(ss * (1.f / 128.f) + EPSLN);
    _Float16* np = &nT[tl * 128 + lane * 4];
    np[0] = (_Float16)(d0 * rs * lg4.x + lb4.x);
    np[1] = (_Float16)(d1 * rs * lg4.y + lb4.y);
    np[2] = (_Float16)(d2 * rs * lg4.z + lb4.z);
    np[3] = (_Float16)(d3 * rs * lg4.w + lb4.w);
  }
  __syncthreads();

  { // b = n @ Wb partials (4 threads per token)
    int tl = threadIdx.x >> 2, part = threadIdx.x & 3;
    float s = 0.f;
    for (int c = part * 32; c < part * 32 + 32; ++c) s += (float)nT[tl * 128 + c] * Wb[c];
    bp[tl * 4 + part] = s;
  }

  // Projections: 32 tiles (4 M x 8 N), 4 tiles per wave, A shared across mats.
#pragma unroll
  for (int ti = 0; ti < 4; ++ti) {
    const int idx = w * 4 + ti;
    const int mt = idx >> 3, nt = idx & 7;
    const v16h a0 = ldA(&nT[mt * 16 * 128 + 0],  lane, 128);
    const v16h a1 = ldA(&nT[mt * 16 * 128 + 32], lane, 128);
    const v16h a2 = ldA(&nT[mt * 16 * 128 + 64], lane, 128);
    const v16h a3 = ldA(&nT[mt * 16 * 128 + 96], lane, 128);
    const int cc = nt * 16 + (lane & 15);
    const int row0 = t0 + mt * 16 + ((lane >> 4) << 3);

    auto proj = [&](int m) -> v8f {
      v8f acc = {};
      acc = wmma16(a0, ldBpk(wPack + ((m * 32 + 0 * 8 + nt) << 9), lane), acc);
      acc = wmma16(a1, ldBpk(wPack + ((m * 32 + 1 * 8 + nt) << 9), lane), acc);
      acc = wmma16(a2, ldBpk(wPack + ((m * 32 + 2 * 8 + nt) << 9), lane), acc);
      acc = wmma16(a3, ldBpk(wPack + ((m * 32 + 3 * 8 + nt) << 9), lane), acc);
      return acc;
    };

    { // q
      v8f acc = proj(0);
#pragma unroll
      for (int r = 0; r < 8; ++r)
        qOut[(size_t)(row0 + r) * 128 + cc] = (_Float16)acc[r];
    }
    { // k
      v8f acc = proj(1);
#pragma unroll
      for (int r = 0; r < 8; ++r)
        kOut[(size_t)(row0 + r) * 128 + cc] = (_Float16)acc[r];
    }
    { // v (stored transposed: vT[i][c][k])
      v8f acc = proj(2);
#pragma unroll
      for (int r = 0; r < 8; ++r) {
        int tok = row0 + r;
        int i = tok / NTOK, kk = tok - i * NTOK;
        vTOut[((size_t)i * 128 + cc) * NTOK + kk] = (_Float16)acc[r];
      }
    }
    { // g = sigmoid(n@Wg + bg)
      v8f acc = proj(3);
      float bgv = bg[cc];
#pragma unroll
      for (int r = 0; r < 8; ++r) {
        float gv = 1.f / (1.f + __expf(-(acc[r] + bgv)));
        gOut[(size_t)(row0 + r) * 128 + cc] = (_Float16)gv;
      }
    }
  }
  __syncthreads();
  if (threadIdx.x < 64)
    bOut[t0 + threadIdx.x] = bp[threadIdx.x * 4] + bp[threadIdx.x * 4 + 1] +
                             bp[threadIdx.x * 4 + 2] + bp[threadIdx.x * 4 + 3];
}

// ---------------------------------------------------------------------------
// Kernel 3: per (row i, 16 k-columns per wave) online softmax stats over j.
// s[j,k] = SCALE*q_j.k_k + b[j]; softmax axis is j => column stats.
// B-operand (the wave's k-tokens) is hoisted out of the j loop.
// grid (3, 384), 256 threads (8 waves).
// ---------------------------------------------------------------------------
__global__ __launch_bounds__(256) void k_stats(const _Float16* __restrict__ q,
                                               const _Float16* __restrict__ kv,
                                               const float* __restrict__ bvec,
                                               float* __restrict__ Mx,
                                               float* __restrict__ Zi) {
  const int i = blockIdx.y, kb = blockIdx.x;
  const int lane = threadIdx.x & 31, w = threadIdx.x >> 5;
  __shared__ float bRow[NTOK];
  for (int t = threadIdx.x; t < NTOK; t += 256) bRow[t] = bvec[i * NTOK + t];
  __syncthreads();

  const int kBase = kb * 128 + w * 16;
  const _Float16* kp = kv + ((size_t)i * NTOK + kBase) * 128;
  const v16h kb0 = ldB(kp + 0,  lane, 128);
  const v16h kb1 = ldB(kp + 32, lane, 128);
  const v16h kb2 = ldB(kp + 64, lane, 128);
  const v16h kb3 = ldB(kp + 96, lane, 128);

  float m = -1e30f, z = 0.f;
  for (int jt = 0; jt < 24; ++jt) {
    const _Float16* qp = q + ((size_t)i * NTOK + jt * 16) * 128;
    v8f acc = {};
    acc = wmma16(ldA(qp + 0,  lane, 128), kb0, acc);
    acc = wmma16(ldA(qp + 32, lane, 128), kb1, acc);
    acc = wmma16(ldA(qp + 64, lane, 128), kb2, acc);
    acc = wmma16(ldA(qp + 96, lane, 128), kb3, acc);
    float s[8], tmax = -1e30f;
#pragma unroll
    for (int r = 0; r < 8; ++r) {
      s[r] = SCALE * acc[r] + bRow[jt * 16 + r + ((lane >> 4) << 3)];
      tmax = fmaxf(tmax, s[r]);
    }
    tmax = fmaxf(tmax, __shfl_xor(tmax, 16));
    float mn = fmaxf(m, tmax);
    float se = 0.f;
#pragma unroll
    for (int r = 0; r < 8; ++r) se += __expf(s[r] - mn);
    se += __shfl_xor(se, 16);
    z = z * __expf(m - mn) + se;
    m = mn;
  }
  if (lane < 16) {
    Mx[i * NTOK + kBase + lane] = m;
    Zi[i * NTOK + kBase + lane] = 1.f / z;
  }
}

// ---------------------------------------------------------------------------
// Kernel 4: O = softmax(S)·V, gate with g, project with Wo (+bo).
// Computes S^T tiles so the normalized-P D-fragment repacks in-lane into the
// next WMMA's A-fragment. The S^T B-operand (the wave's q-tokens) is hoisted
// out of the k loop. grid (6, 384), 128 threads (4 waves, 16 j each).
// ---------------------------------------------------------------------------
__global__ __launch_bounds__(128) void k_attn_out(const _Float16* __restrict__ q,
                                                  const _Float16* __restrict__ kv,
                                                  const _Float16* __restrict__ g,
                                                  const _Float16* __restrict__ vT,
                                                  const float* __restrict__ bvec,
                                                  const float* __restrict__ Mx,
                                                  const float* __restrict__ Zi,
                                                  const _Float16* __restrict__ wPack,
                                                  const float* __restrict__ bo,
                                                  float* __restrict__ out) {
  const int i = blockIdx.y, jb = blockIdx.x;
  const int lane = threadIdx.x & 31, w = threadIdx.x >> 5;
  __shared__ float sM[NTOK], sZ[NTOK];
  __shared__ _Float16 sG[64 * 128];
  __shared__ _Float16 sH[4][16 * 128];

  for (int t = threadIdx.x; t < NTOK; t += 128) {
    sM[t] = Mx[i * NTOK + t];
    sZ[t] = Zi[i * NTOK + t];
  }
  {
    const v8h* src = (const v8h*)(g + ((size_t)i * NTOK + jb * 64) * 128);
    v8h* dst = (v8h*)sG;
    for (int t = threadIdx.x; t < 1024; t += 128) dst[t] = src[t];
  }
  __syncthreads();

  const int jW = jb * 64 + w * 16;
  const float bj = bvec[i * NTOK + jW + (lane & 15)];
  const _Float16* qp = q + ((size_t)i * NTOK + jW) * 128;
  const v16h qb0 = ldB(qp + 0,  lane, 128);
  const v16h qb1 = ldB(qp + 32, lane, 128);
  const v16h qb2 = ldB(qp + 64, lane, 128);
  const v16h qb3 = ldB(qp + 96, lane, 128);

  v8f o[8] = {};
  for (int ks = 0; ks < NTOK; ks += 32) {
    HF16 P;
#pragma unroll
    for (int hb = 0; hb < 2; ++hb) {
      int k16 = ks + hb * 16;
      const _Float16* kp = kv + ((size_t)i * NTOK + k16) * 128;
      v8f acc = {};
      acc = wmma16(ldA(kp + 0,  lane, 128), qb0, acc);  // S^T: A = k-tokens
      acc = wmma16(ldA(kp + 32, lane, 128), qb1, acc);
      acc = wmma16(ldA(kp + 64, lane, 128), qb2, acc);
      acc = wmma16(ldA(kp + 96, lane, 128), qb3, acc);
#pragma unroll
      for (int r = 0; r < 8; ++r) {
        int kr = k16 + r + ((lane >> 4) << 3);
        float p = __expf(SCALE * acc[r] + bj - sM[kr]) * sZ[kr];
        P.e[hb * 8 + r] = (_Float16)p;  // in-lane D->A repack
      }
    }
#pragma unroll
    for (int ct = 0; ct < 8; ++ct) {
      v16h bv = ldB(vT + ((size_t)i * 128 + ct * 16) * NTOK + ks, lane, NTOK);
      o[ct] = wmma16(P.v, bv, o[ct]);
    }
  }

  // gate with g, stage f16 tile in LDS for the Wo GEMM
#pragma unroll
  for (int ct = 0; ct < 8; ++ct) {
    int cc = ct * 16 + (lane & 15);
#pragma unroll
    for (int r = 0; r < 8; ++r) {
      int rl = r + ((lane >> 4) << 3);
      float hv = (float)sG[(w * 16 + rl) * 128 + cc] * o[ct][r];
      sH[w][rl * 128 + cc] = (_Float16)hv;
    }
  }
  // output projection: (g*O) @ Wo + bo
#pragma unroll
  for (int nt = 0; nt < 8; ++nt) {
    v8f acc = {};
#pragma unroll
    for (int kb = 0; kb < 4; ++kb)
      acc = wmma16(ldA(&sH[w][kb * 32], lane, 128),
                   ldBpk(wPack + ((4 * 32 + kb * 8 + nt) << 9), lane), acc);
    int cc = nt * 16 + (lane & 15);
    float bov = bo[cc];
#pragma unroll
    for (int r = 0; r < 8; ++r) {
      int j = jW + r + ((lane >> 4) << 3);
      out[((size_t)i * NTOK + j) * 128 + cc] = acc[r] + bov;
    }
  }
}

// ---------------------------------------------------------------------------
extern "C" void kernel_launch(void* const* d_in, const int* in_sizes, int n_in,
                              void* d_out, int out_size, void* d_ws, size_t ws_size,
                              hipStream_t stream) {
  const float* x   = (const float*)d_in[0];
  const float* lng = (const float*)d_in[1];
  const float* lnb = (const float*)d_in[2];
  const float* Wq  = (const float*)d_in[3];
  const float* Wk  = (const float*)d_in[4];
  const float* Wv  = (const float*)d_in[5];
  const float* Wb  = (const float*)d_in[6];
  const float* Wg  = (const float*)d_in[7];
  const float* bg  = (const float*)d_in[8];
  const float* Wo  = (const float*)d_in[9];
  const float* bo  = (const float*)d_in[10];
  float* out = (float*)d_out;

  char* ws = (char*)d_ws;
  size_t off = 0;
  auto carve = [&](size_t bytes) { size_t o = off; off = (off + bytes + 255) & ~(size_t)255; return o; };
  const size_t fBytes = (size_t)TTOK * 128 * sizeof(_Float16);  // 37.75 MB
  size_t wPackOff = carve(5 * 128 * 128 * sizeof(_Float16));
  size_t qOff  = carve(fBytes);
  size_t kOff  = carve(fBytes);
  size_t gOff  = carve(fBytes);
  size_t vTOff = carve(fBytes);
  size_t bOff  = carve((size_t)TTOK * sizeof(float));
  size_t mxOff = carve((size_t)TTOK * sizeof(float));
  size_t zOff  = carve((size_t)TTOK * sizeof(float));
  (void)ws_size; (void)n_in; (void)in_sizes; (void)out_size;

  _Float16* wPack = (_Float16*)(ws + wPackOff);
  _Float16* qB  = (_Float16*)(ws + qOff);
  _Float16* kB  = (_Float16*)(ws + kOff);
  _Float16* gB  = (_Float16*)(ws + gOff);
  _Float16* vTB = (_Float16*)(ws + vTOff);
  float* bV  = (float*)(ws + bOff);
  float* MxB = (float*)(ws + mxOff);
  float* ZiB = (float*)(ws + zOff);

  k_pack<<<(5 * 32 * 32 * 16 + 255) / 256, 256, 0, stream>>>(Wq, Wk, Wv, Wg, Wo, wPack);
  k_lnproj<<<TTOK / 64, 256, 0, stream>>>(x, lng, lnb, Wb, bg, wPack, qB, kB, gB, vTB, bV);
  k_stats<<<dim3(3, NTOK), 256, 0, stream>>>(qB, kB, bV, MxB, ZiB);
  k_attn_out<<<dim3(6, NTOK), 128, 0, stream>>>(qB, kB, gB, vTB, bV, MxB, ZiB, wPack, bo, out);
}
